// VGGNN_49589692399794
// MI455X (gfx1250) — matrix operand order repaired and checked by reading, not compiled, not run on hardware
//
#include <hip/hip_runtime.h>
#include <hip/hip_bf16.h>

// ---------------------------------------------------------------------------
// VGAE with ChebConv (K=2) on MI455X / gfx1250, wave32 + WMMA bf16.
// ---------------------------------------------------------------------------

#define NN   16384
#define HH   128
#define ZZ   32
#define EE   524288

typedef __attribute__((ext_vector_type(16))) __bf16 v16bf;
typedef __attribute__((ext_vector_type(8)))  __bf16 v8bf;
typedef __attribute__((ext_vector_type(8)))  float  v8f;

// ---- WMMA fragment helpers (layouts per CDNA5 ISA 7.12.2) ------------------

// A fragment, 16x32 bf16 (M x K). row = base pointer of this lane's row
// (row index = lane & 15) at K=0 of the current k-step.
__device__ __forceinline__ v16bf frag_a(const __bf16* row, int lane) {
    const int base = (lane >= 16) ? 8 : 0;
    v8bf lo = *(const v8bf*)(row + base);        // K = base .. base+7
    v8bf hi = *(const v8bf*)(row + base + 16);   // K = base+16 .. base+23
    return __builtin_shufflevector(lo, hi, 0,1,2,3,4,5,6,7,8,9,10,11,12,13,14,15);
}

// B fragment, 32x16 bf16 (K x N), staged in LDS transposed as [N][K].
// row = base pointer of this lane's column-row (index = lane & 15) at K=0.
__device__ __forceinline__ v16bf frag_b(const __bf16* row, int lane) {
    const int base = (lane >= 16) ? 8 : 0;
    v8bf lo = *(const v8bf*)(row + base);        // K = base .. base+7
    v8bf hi = *(const v8bf*)(row + base + 16);   // K = base+16 .. base+23
    return __builtin_shufflevector(lo, hi, 0,1,2,3,4,5,6,7,8,9,10,11,12,13,14,15);
}

__device__ __forceinline__ v8f wmma_bf16(v16bf a, v16bf b, v8f c) {
    return __builtin_amdgcn_wmma_f32_16x16x32_bf16(
        /*neg_a=*/false, a, /*neg_b=*/false, b,
        /*c_mod=*/(short)0, c, /*reuse_a=*/false, /*reuse_b=*/false);
}

__device__ __forceinline__ void cvt16(__bf16* dst, float4 f0, float4 f1,
                                      float4 f2, float4 f3) {
    dst[0]=(__bf16)f0.x; dst[1]=(__bf16)f0.y; dst[2]=(__bf16)f0.z; dst[3]=(__bf16)f0.w;
    dst[4]=(__bf16)f1.x; dst[5]=(__bf16)f1.y; dst[6]=(__bf16)f1.z; dst[7]=(__bf16)f1.w;
    dst[8]=(__bf16)f2.x; dst[9]=(__bf16)f2.y; dst[10]=(__bf16)f2.z; dst[11]=(__bf16)f2.w;
    dst[12]=(__bf16)f3.x; dst[13]=(__bf16)f3.y; dst[14]=(__bf16)f3.z; dst[15]=(__bf16)f3.w;
}

// ---------------------------------------------------------------------------
// Kernel 1: h = relu(x_adj @ Wt + bt)     [N,N] x [N,H] -> [N,H]
// Grid: N/128 blocks x 256 threads (8 waves). Block computes 128 rows x 128
// cols. Wave tiling 2x4: wave (wr,wc) owns 32 rows x 64 cols -> per k-step
// 2 A-frags + 4 B-frags feed 8 WMMAs; each B-frag is reused by two
// back-to-back WMMAs so LDS-load waits overlap matrix ops.
// K-tile = 64, double-buffered LDS; next-tile global loads issued before the
// WMMA chain; global_prefetch two tiles ahead.
// ---------------------------------------------------------------------------
__global__ __launch_bounds__(256) void gemm1_kernel(
        const float* __restrict__ A,    // x_adj [N][N]
        const float* __restrict__ Wt,   // [N][H]
        const float* __restrict__ bt,   // [H]
        float* __restrict__ hOut)       // [N][H]
{
    constexpr int KT = 64;              // k-tile
    constexpr int LD = 72;              // KT + pad, 144B rows (16B aligned)
    __shared__ __bf16 lA[2][128][LD];   // [buf][row][k]
    __shared__ __bf16 lB[2][128][LD];   // [buf][n][k]  (transposed Wt chunk)

    const int t    = threadIdx.x;
    const int lane = t & 31;
    const int wave = t >> 5;
    const int rowBase = blockIdx.x * 128;

    const int wr = (wave >> 1) * 32;    // wave row offset: 0,32,64,96
    const int wc = (wave & 1) * 64;     // wave col offset: 0,64

    v8f acc[8] = {};                    // acc[2*ct + rt]

    // staging assignments: 32 contiguous floats per thread per matrix
    const int arow = t >> 1;            // 0..127
    const int aseg = (t & 1) * 32;      // 0 / 32
    const int bk   = t >> 2;            // 0..63
    const int bn   = (t & 3) * 32;      // 0..96

    const float* aSrc = A  + (size_t)(rowBase + arow) * NN + aseg;
    const float* bSrc = Wt + (size_t)bk * HH + bn;

    // ---- prologue: stage tile 0 into buffer 0 -----------------------------
    {
        float4 sa[8], sb[8];
#pragma unroll
        for (int i = 0; i < 8; ++i) sa[i] = ((const float4*)aSrc)[i];
#pragma unroll
        for (int i = 0; i < 8; ++i) sb[i] = ((const float4*)bSrc)[i];
        cvt16(&lA[0][arow][aseg +  0], sa[0], sa[1], sa[2], sa[3]);
        cvt16(&lA[0][arow][aseg + 16], sa[4], sa[5], sa[6], sa[7]);
#pragma unroll
        for (int i = 0; i < 8; ++i) {
            lB[0][bn + 4*i + 0][bk] = (__bf16)sb[i].x;
            lB[0][bn + 4*i + 1][bk] = (__bf16)sb[i].y;
            lB[0][bn + 4*i + 2][bk] = (__bf16)sb[i].z;
            lB[0][bn + 4*i + 3][bk] = (__bf16)sb[i].w;
        }
    }
    __syncthreads();

    int cur = 0;
    for (int k0 = 0; k0 < NN; k0 += KT) {
        const bool hasNext = (k0 + KT) < NN;
        float4 sa[8], sb[8];
        if (hasNext) {                   // issue next-tile global loads now
            const float4* ap = (const float4*)(aSrc + k0 + KT);
            const float4* bp = (const float4*)(bSrc + (size_t)(k0 + KT) * HH);
#pragma unroll
            for (int i = 0; i < 8; ++i) sa[i] = ap[i];
#pragma unroll
            for (int i = 0; i < 8; ++i) sb[i] = bp[i];
        }
        if (k0 + 2 * KT < NN) {          // L2 prefetch two tiles ahead
            __builtin_prefetch(aSrc + k0 + 2 * KT, 0, 1);
            __builtin_prefetch(bSrc + (size_t)(k0 + 2 * KT) * HH, 0, 1);
        }

        // ---- compute: 2 k-steps, 2 row tiles x 4 col tiles ----------------
#pragma unroll
        for (int ks = 0; ks < 2; ++ks) {
            v16bf af0 = frag_a(&lA[cur][wr +      (lane & 15)][ks * 32], lane);
            v16bf af1 = frag_a(&lA[cur][wr + 16 + (lane & 15)][ks * 32], lane);
#pragma unroll
            for (int ct = 0; ct < 4; ++ct) {
                v16bf bf = frag_b(&lB[cur][wc + ct * 16 + (lane & 15)][ks * 32], lane);
                acc[2*ct + 0] = wmma_bf16(af0, bf, acc[2*ct + 0]);
                acc[2*ct + 1] = wmma_bf16(af1, bf, acc[2*ct + 1]);
            }
        }

        // ---- stage next tile into the other buffer ------------------------
        if (hasNext) {
            const int nxt = cur ^ 1;
            cvt16(&lA[nxt][arow][aseg +  0], sa[0], sa[1], sa[2], sa[3]);
            cvt16(&lA[nxt][arow][aseg + 16], sa[4], sa[5], sa[6], sa[7]);
#pragma unroll
            for (int i = 0; i < 8; ++i) {
                lB[nxt][bn + 4*i + 0][bk] = (__bf16)sb[i].x;
                lB[nxt][bn + 4*i + 1][bk] = (__bf16)sb[i].y;
                lB[nxt][bn + 4*i + 2][bk] = (__bf16)sb[i].z;
                lB[nxt][bn + 4*i + 3][bk] = (__bf16)sb[i].w;
            }
        }
        __syncthreads();
        cur ^= 1;
    }

    // ---- epilogue: bias + relu -------------------------------------------
    const int col0 = lane & 15;
    const int mOff = (lane >= 16) ? 8 : 0;
#pragma unroll
    for (int ct = 0; ct < 4; ++ct) {
        const int col = wc + ct * 16 + col0;
        const float bias = bt[col];
#pragma unroll
        for (int rt = 0; rt < 2; ++rt) {
            const int r0 = rowBase + wr + rt * 16 + mOff;
            v8f a = acc[2*ct + rt];
#pragma unroll
            for (int v = 0; v < 8; ++v) {
                float val = fmaxf(a[v] + bias, 0.0f);
                hOut[(size_t)(r0 + v) * HH + col] = val;
            }
        }
    }
}

// ---------------------------------------------------------------------------
// Degree + D^{-1/2}
// ---------------------------------------------------------------------------
__global__ void degree_kernel(const long long* __restrict__ src,
                              float* __restrict__ deg, int e) {
    int i = blockIdx.x * blockDim.x + threadIdx.x;
    if (i < e) atomicAdd(&deg[(int)src[i]], 1.0f);
}

__global__ void dinv_kernel(const float* __restrict__ deg,
                            float* __restrict__ dinv, int n) {
    int i = blockIdx.x * blockDim.x + threadIdx.x;
    if (i < n) {
        float d = deg[i];
        dinv[i] = (d > 0.0f) ? rsqrtf(d) : 0.0f;
    }
}

// ---------------------------------------------------------------------------
// SpMM: out[dst] += -(dinv[src]*dinv[dst]) * x[src]   (F features)
// One wave per edge iteration; lane handles features lane, lane+32, ...
// ---------------------------------------------------------------------------
template <int F>
__global__ void spmm_kernel(const long long* __restrict__ eidx,
                            const float* __restrict__ dinv,
                            const float* __restrict__ x,
                            float* __restrict__ out, int e) {
    const long long* src = eidx;
    const long long* dst = eidx + e;
    const int lane = threadIdx.x & 31;
    const int wave = blockIdx.x * (blockDim.x >> 5) + (threadIdx.x >> 5);
    const int nw   = gridDim.x * (blockDim.x >> 5);
    for (int ed = wave; ed < e; ed += nw) {
        int s = (int)src[ed], d = (int)dst[ed];
        float w = -(dinv[s] * dinv[d]);
#pragma unroll
        for (int f = lane; f < F; f += 32)
            atomicAdd(&out[(size_t)d * F + f], w * x[(size_t)s * F + f]);
    }
}

// ---------------------------------------------------------------------------
// Kernel: h1 = relu(h @ W1a + tx1 @ W1b + b1)   [N,128]x[128,64] -> [N,64]
// Grid: N/32 blocks x 256 threads; wave = (rowTile 0..1, colTile 0..3).
// ---------------------------------------------------------------------------
__global__ __launch_bounds__(256) void layer1_kernel(
        const float* __restrict__ h, const float* __restrict__ tx1,
        const float* __restrict__ W1a, const float* __restrict__ W1b,
        const float* __restrict__ b1, float* __restrict__ h1)
{
    constexpr int LD = 136;             // 128 K + pad (272B rows)
    __shared__ __bf16 lH[32][LD];
    __shared__ __bf16 lT[32][LD];
    __shared__ __bf16 lWa[64][LD];      // [n][k]
    __shared__ __bf16 lWb[64][LD];

    const int t = threadIdx.x, lane = t & 31, wave = t >> 5;
    const int rowBase = blockIdx.x * 32;

    {   // stage activations (32 x 128 each)
        int r = t >> 3, c0 = (t & 7) * 16;
        const float4* hp = (const float4*)(h   + (size_t)(rowBase + r) * HH + c0);
        const float4* tp = (const float4*)(tx1 + (size_t)(rowBase + r) * HH + c0);
        cvt16(&lH[r][c0], hp[0], hp[1], hp[2], hp[3]);
        cvt16(&lT[r][c0], tp[0], tp[1], tp[2], tp[3]);
    }
    for (int i = t; i < 128 * 64; i += 256) {   // stage weights transposed
        int k = i >> 6, n = i & 63;
        lWa[n][k] = (__bf16)W1a[i];
        lWb[n][k] = (__bf16)W1b[i];
    }
    __syncthreads();

    const int rowTile = wave >> 2, colTile = wave & 3;
    const __bf16* arp = &lH[rowTile * 16 + (lane & 15)][0];
    const __bf16* trp = &lT[rowTile * 16 + (lane & 15)][0];
    const __bf16* war = &lWa[colTile * 16 + (lane & 15)][0];
    const __bf16* wbr = &lWb[colTile * 16 + (lane & 15)][0];

    v8f acc = {};
#pragma unroll
    for (int k0 = 0; k0 < 128; k0 += 32)
        acc = wmma_bf16(frag_a(arp + k0, lane), frag_b(war + k0, lane), acc);
#pragma unroll
    for (int k0 = 0; k0 < 128; k0 += 32)
        acc = wmma_bf16(frag_a(trp + k0, lane), frag_b(wbr + k0, lane), acc);

    const int col = colTile * 16 + (lane & 15);
    const float bias = b1[col];
    const int r0 = rowBase + rowTile * 16 + ((lane >= 16) ? 8 : 0);
#pragma unroll
    for (int v = 0; v < 8; ++v)
        h1[(size_t)(r0 + v) * 64 + col] = fmaxf(acc[v] + bias, 0.0f);
}

// ---------------------------------------------------------------------------
// Kernel: mu/logstd/z + KL partial.
//   mu     = h1 @ Wma + tx2 @ Wmb + bm
//   logstd = min(h1 @ Wla + tx2 @ Wlb + bl, 10)
//   z      = mu + eps * exp(logstd)        -> d_out
//   klSum += sum(1 + 2*logstd - mu^2 - exp(logstd)^2)
// Grid: N/64 blocks x 256 threads; wave = (rowTile 0..3, colTile 0..1).
// ---------------------------------------------------------------------------
__global__ __launch_bounds__(256) void layer2_kernel(
        const float* __restrict__ h1, const float* __restrict__ tx2,
        const float* __restrict__ Wma, const float* __restrict__ Wmb,
        const float* __restrict__ bm,
        const float* __restrict__ Wla, const float* __restrict__ Wlb,
        const float* __restrict__ bl,
        const float* __restrict__ eps,
        float* __restrict__ zOut, float* __restrict__ klSum)
{
    constexpr int LD = 72;              // 64 K + pad (144B rows)
    __shared__ __bf16 lH[64][LD];
    __shared__ __bf16 lT[64][LD];
    __shared__ __bf16 lWma[32][LD], lWmb[32][LD], lWla[32][LD], lWlb[32][LD];

    const int t = threadIdx.x, lane = t & 31, wave = t >> 5;
    const int rowBase = blockIdx.x * 64;

    {   // stage activations (64 x 64 each)
        int r = t >> 2, c0 = (t & 3) * 16;
        const float4* hp = (const float4*)(h1  + (size_t)(rowBase + r) * 64 + c0);
        const float4* tp = (const float4*)(tx2 + (size_t)(rowBase + r) * 64 + c0);
        cvt16(&lH[r][c0], hp[0], hp[1], hp[2], hp[3]);
        cvt16(&lT[r][c0], tp[0], tp[1], tp[2], tp[3]);
    }
    for (int i = t; i < 64 * 32; i += 256) {    // stage weights transposed
        int k = i >> 5, n = i & 31;
        lWma[n][k] = (__bf16)Wma[i];
        lWmb[n][k] = (__bf16)Wmb[i];
        lWla[n][k] = (__bf16)Wla[i];
        lWlb[n][k] = (__bf16)Wlb[i];
    }
    __syncthreads();

    const int rowTile = wave >> 1, colTile = wave & 1;
    const __bf16* arp = &lH[rowTile * 16 + (lane & 15)][0];
    const __bf16* trp = &lT[rowTile * 16 + (lane & 15)][0];
    const int cn = colTile * 16 + (lane & 15);

    v8f aMu = {}, aLs = {};
#pragma unroll
    for (int k0 = 0; k0 < 64; k0 += 32) {
        v16bf ah = frag_a(arp + k0, lane);
        aMu = wmma_bf16(ah, frag_b(&lWma[cn][k0], lane), aMu);
        aLs = wmma_bf16(ah, frag_b(&lWla[cn][k0], lane), aLs);
    }
#pragma unroll
    for (int k0 = 0; k0 < 64; k0 += 32) {
        v16bf at = frag_a(trp + k0, lane);
        aMu = wmma_bf16(at, frag_b(&lWmb[cn][k0], lane), aMu);
        aLs = wmma_bf16(at, frag_b(&lWlb[cn][k0], lane), aLs);
    }

    const int col = colTile * 16 + (lane & 15);
    const float bmv = bm[col], blv = bl[col];
    const int r0 = rowBase + rowTile * 16 + ((lane >= 16) ? 8 : 0);
    float kl = 0.0f;
#pragma unroll
    for (int v = 0; v < 8; ++v) {
        float mu = aMu[v] + bmv;
        float ls = fminf(aLs[v] + blv, 10.0f);
        float el = expf(ls);
        float zv = mu + eps[(size_t)(r0 + v) * ZZ + col] * el;
        zOut[(size_t)(r0 + v) * ZZ + col] = zv;
        kl += 1.0f + 2.0f * ls - mu * mu - el * el;
    }
#pragma unroll
    for (int o = 16; o > 0; o >>= 1) kl += __shfl_xor(kl, o, 32);
    if (lane == 0) atomicAdd(klSum, kl);
}

// ---------------------------------------------------------------------------
// Edge reconstruction loss: one wave per edge, lane = feature (Z == 32).
// ---------------------------------------------------------------------------
__global__ void edge_loss_kernel(const long long* __restrict__ eidx,
                                 const float* __restrict__ z,
                                 float* __restrict__ accum, int e, int isNeg) {
    const long long* src = eidx;
    const long long* dst = eidx + e;
    const int lane = threadIdx.x & 31;
    const int wave = blockIdx.x * (blockDim.x >> 5) + (threadIdx.x >> 5);
    const int nw   = gridDim.x * (blockDim.x >> 5);
    float local = 0.0f;
    for (int ed = wave; ed < e; ed += nw) {
        int s = (int)src[ed], d = (int)dst[ed];
        float p = z[(size_t)s * ZZ + lane] * z[(size_t)d * ZZ + lane];
#pragma unroll
        for (int o = 16; o > 0; o >>= 1) p += __shfl_xor(p, o, 32);
        float sgm = 1.0f / (1.0f + expf(-p));
        local += isNeg ? -logf(1.0f - sgm + 1e-15f) : -logf(sgm + 1e-15f);
    }
    if (lane == 0) atomicAdd(accum, local);
}

__global__ void finalize_kernel(const float* __restrict__ sums,
                                float* __restrict__ out) {
    float posL = sums[0] * (1.0f / (float)EE);
    float negL = sums[1] * (1.0f / (float)EE);
    float kl   = -0.5f * sums[2] / (float)NN;
    out[(size_t)NN * ZZ] = posL + negL + kl / (float)NN;
}

// ---------------------------------------------------------------------------
// Workspace layout (bytes)
// ---------------------------------------------------------------------------
#define OFF_TX1   ((size_t)0)                       // N*128*4 = 8 MB (zeroed)
#define OFF_TX2   ((size_t)8388608)                 // N*64*4  = 4 MB (zeroed)
#define OFF_DEG   ((size_t)12582912)                // N*4              (zeroed)
#define OFF_SUMS  ((size_t)12648448)                // 3 floats         (zeroed)
#define OFF_DINV  ((size_t)12648704)                // N*4
#define OFF_H     ((size_t)12714240)                // N*128*4 = 8 MB
#define OFF_H1    ((size_t)21102848)                // N*64*4  = 4 MB
#define ZERO_BYTES OFF_DINV

extern "C" void kernel_launch(void* const* d_in, const int* in_sizes, int n_in,
                              void* d_out, int out_size, void* d_ws, size_t ws_size,
                              hipStream_t stream) {
    const float* x_adj = (const float*)d_in[0];
    const float* Wt    = (const float*)d_in[1];
    const float* bt    = (const float*)d_in[2];
    const float* W1a   = (const float*)d_in[3];
    const float* W1b   = (const float*)d_in[4];
    const float* b1    = (const float*)d_in[5];
    const float* Wma   = (const float*)d_in[6];
    const float* Wmb   = (const float*)d_in[7];
    const float* bm    = (const float*)d_in[8];
    const float* Wla   = (const float*)d_in[9];
    const float* Wlb   = (const float*)d_in[10];
    const float* bl    = (const float*)d_in[11];
    const float* eps   = (const float*)d_in[12];
    const long long* ei  = (const long long*)d_in[13];
    const long long* nei = (const long long*)d_in[14];
    float* out = (float*)d_out;

    char* ws   = (char*)d_ws;
    float* tx1  = (float*)(ws + OFF_TX1);
    float* tx2  = (float*)(ws + OFF_TX2);
    float* deg  = (float*)(ws + OFF_DEG);
    float* sums = (float*)(ws + OFF_SUMS);   // [pos, neg, kl]
    float* dinv = (float*)(ws + OFF_DINV);
    float* hbuf = (float*)(ws + OFF_H);
    float* h1b  = (float*)(ws + OFF_H1);

    hipMemsetAsync(ws, 0, ZERO_BYTES, stream);

    // encoder trunk: big bf16 WMMA GEMM
    gemm1_kernel<<<NN / 128, 256, 0, stream>>>(x_adj, Wt, bt, hbuf);

    // graph normalization
    degree_kernel<<<EE / 256, 256, 0, stream>>>(ei, deg, EE);
    dinv_kernel<<<NN / 256, 256, 0, stream>>>(deg, dinv, NN);

    // ChebConv layer 1
    spmm_kernel<128><<<2048, 256, 0, stream>>>(ei, dinv, hbuf, tx1, EE);
    layer1_kernel<<<NN / 32, 256, 0, stream>>>(hbuf, tx1, W1a, W1b, b1, h1b);

    // ChebConv layer 2 (shared T1 propagation for mu and logstd)
    spmm_kernel<64><<<2048, 256, 0, stream>>>(ei, dinv, h1b, tx2, EE);
    layer2_kernel<<<NN / 64, 256, 0, stream>>>(h1b, tx2, Wma, Wmb, bm,
                                               Wla, Wlb, bl, eps, out, &sums[2]);

    // decoder losses
    edge_loss_kernel<<<1024, 256, 0, stream>>>(ei,  out, &sums[0], EE, 0);
    edge_loss_kernel<<<1024, 256, 0, stream>>>(nei, out, &sums[1], EE, 1);
    finalize_kernel<<<1, 1, 0, stream>>>(sums, out);
}
